// EnhancedMaskDecoder_40690520162367
// MI455X (gfx1250) — matrix operand order repaired and checked by reading, not compiled, not run on hardware
//
#include <hip/hip_runtime.h>
#include <math.h>

typedef __bf16 bf16_t;
typedef __attribute__((ext_vector_type(16))) __bf16 v16bf;
typedef __attribute__((ext_vector_type(4)))  __bf16 v4bf;
typedef __attribute__((ext_vector_type(8)))  float  v8f;
typedef __attribute__((ext_vector_type(4)))  float  v4f;

#define BB_     8
#define L_      1024
#define H_      1024
#define VOCAB_  50257
#define EVOCAB_ 64
#define LN_EPS_ 1e-7f
#define LASSO_  1e-4f
#define CAPR_   8192          // max rows per GEMM chunk
#define ECAP_   20480         // max nz edges
#define E2CAP_  40960         // max total edges (nz + transposed)

// ---------------- ordered compaction of masked token positions ----------------
__global__ __launch_bounds__(1024)
void k_compact_mask(const int* __restrict__ tflat, int* mask_idx, int* lm_lab, int* counts) {
  __shared__ int sc[1024];
  int t = threadIdx.x;
  int base = t * 8;
  int loc[8]; int cnt = 0;
  #pragma unroll
  for (int i = 0; i < 8; ++i) { int v = tflat[base + i]; loc[i] = v; cnt += (v > 0); }
  sc[t] = cnt; __syncthreads();
  for (int off = 1; off < 1024; off <<= 1) {
    int v = (t >= off) ? sc[t - off] : 0; __syncthreads();
    sc[t] += v; __syncthreads();
  }
  int pos = sc[t] - cnt;
  #pragma unroll
  for (int i = 0; i < 8; ++i)
    if (loc[i] > 0) { mask_idx[pos] = base + i; lm_lab[pos] = loc[i]; ++pos; }
  if (t == 1023) counts[0] = sc[1023];
}

// ---------------- edge list: count / scan / ordered emit ----------------
__global__ __launch_bounds__(256)
void k_edge_count(const int* __restrict__ adj, int* blkCnt) {
  int t = threadIdx.x;
  long long base = (long long)blockIdx.x * 8192;
  int cnt = 0;
  #pragma unroll 4
  for (int i = 0; i < 32; ++i) cnt += (adj[base + (long long)i * 256 + t] != 0);
  __shared__ int sc[256];
  sc[t] = cnt; __syncthreads();
  for (int off = 128; off > 0; off >>= 1) { if (t < off) sc[t] += sc[t + off]; __syncthreads(); }
  if (t == 0) blkCnt[blockIdx.x] = sc[0];
}

__global__ __launch_bounds__(1024)
void k_edge_scan(const int* __restrict__ blkCnt, int* blkOff, int* counts) {
  __shared__ int sc[1024];
  int t = threadIdx.x;
  int c = blkCnt[t]; sc[t] = c; __syncthreads();
  for (int off = 1; off < 1024; off <<= 1) {
    int v = (t >= off) ? sc[t - off] : 0; __syncthreads();
    sc[t] += v; __syncthreads();
  }
  blkOff[t] = sc[t] - c;
  if (t == 1023) counts[1] = sc[1023];
}

__global__ __launch_bounds__(256)
void k_edge_emit(const int* __restrict__ adj, const int* __restrict__ blkOff,
                 int* e_b, int* e_i, int* e_j, int* r0, int* r1, int* lab) {
  int t = threadIdx.x;
  long long base = (long long)blockIdx.x * 8192 + (long long)t * 32;
  int cnt = 0;
  for (int i = 0; i < 32; ++i) cnt += (adj[base + i] != 0);
  __shared__ int sc[256];
  sc[t] = cnt; __syncthreads();
  for (int off = 1; off < 256; off <<= 1) {
    int v = (t >= off) ? sc[t - off] : 0; __syncthreads();
    sc[t] += v; __syncthreads();
  }
  int pos = blkOff[blockIdx.x] + sc[t] - cnt;
  for (int i = 0; i < 32; ++i) {
    long long f = base + i;
    int v = adj[f];
    if (v != 0) {
      int b  = (int)(f >> 20);                    // / (L*L), L*L = 2^20
      int rm = (int)(f & ((1 << 20) - 1));
      int ii = rm >> 10, jj = rm & 1023;
      e_b[pos] = b; e_i[pos] = ii; e_j[pos] = jj;
      r0[pos] = b * L_ + jj;                      // edge_to   = hidden[b, j]
      r1[pos] = b * L_ + ii;                      // edge_from = hidden[b, i]
      lab[pos] = v;
      ++pos;
    }
  }
}

__global__ __launch_bounds__(1024)
void k_trans_emit(const int* __restrict__ adj,
                  const int* __restrict__ e_b, const int* __restrict__ e_i, const int* __restrict__ e_j,
                  int* r0, int* r1, int* lab, int* counts) {
  __shared__ int sc[1024];
  int t = threadIdx.x;
  int Nnz = counts[1];
  int q = (Nnz + 1023) >> 10;
  int s = t * q, e = s + q; if (e > Nnz) e = Nnz; if (s > Nnz) s = Nnz;
  int cnt = 0;
  for (int p = s; p < e; ++p) cnt += (e_i[p] != e_j[p]);
  sc[t] = cnt; __syncthreads();
  for (int off = 1; off < 1024; off <<= 1) {
    int v = (t >= off) ? sc[t - off] : 0; __syncthreads();
    sc[t] += v; __syncthreads();
  }
  int pos = Nnz + sc[t] - cnt;
  for (int p = s; p < e; ++p) {
    if (e_i[p] != e_j[p]) {
      int b = e_b[p], ii = e_i[p], jj = e_j[p];
      r0[pos] = b * L_ + ii;                      // trans: to = hidden[b, i]
      r1[pos] = b * L_ + jj;                      // trans: from = hidden[b, j]
      lab[pos] = adj[((long long)b * L_ + jj) * L_ + ii];   // adj[b, j, i]
      ++pos;
    }
  }
  if (t == 1023) { counts[2] = sc[1023]; counts[3] = Nnz + sc[1023]; }
}

// ---------------- WMMA bf16 GEMM: C[m,n] = sum_k A[m,k]*B[n,k] + bias[n] ----------------
// Block tile 64M x 128N, 8 waves as 2x4, wave tile 32x32 (2x2 WMMA tiles).
// AMODE 0: A fp32, rows gathered via rowmap       (lm dense)
// AMODE 1: A fp32, concat-gather via r0/r1 tables (gm dense, K = 2*H)
// AMODE 2: A bf16, direct                         (logits GEMMs)
// GMOUT:   write into d_out at device-computed gm_logits offset
template <int AMODE, bool GMOUT>
__global__ __launch_bounds__(256)
void k_gemm(const float* __restrict__ Af, const bf16_t* __restrict__ Ab,
            const float* __restrict__ Bf, const float* __restrict__ bias,
            float* __restrict__ Cp, long long ldc,
            const int* __restrict__ rowmap, const int* __restrict__ r0t, const int* __restrict__ r1t,
            const int* __restrict__ counts, int mIdx, int Moff, int N, int K) {
  int M = counts[mIdx] - Moff;
  if (M > CAPR_) M = CAPR_;
  if ((int)blockIdx.y * 64 >= M) return;      // uniform exit (covers M<=0)

  // stride 36 halves = 72 B rows: 8B-aligned quad stores, 18-word stride is
  // conflict-free across 16 lanes (gcd(18,64)=2, 18*r mod 64 all distinct).
  __shared__ __align__(16) bf16_t As[64][36];
  __shared__ __align__(16) bf16_t Bs[128][36];

  int t = threadIdx.x;
  int bm = blockIdx.y * 64, bn = blockIdx.x * 128;
  int lane = t & 31, wave = t >> 5;
  int wrow = wave >> 2, wcol = wave & 3;      // 2x4 waves
  int hf = lane >> 4;

  v8f acc[2][2] = {};

  for (int k0 = 0; k0 < K; k0 += 32) {
    // stage A tile (64 rows x 32 k) as quads: b128 load -> pk bf16 -> b64 LDS store
    #pragma unroll
    for (int i = 0; i < 2; ++i) {
      int g = i * 256 + t;                    // 512 quad groups
      int row = g >> 3;
      int col = (g & 7) * 4;
      int m = bm + row;
      int mc = (m < M) ? m : (M - 1);
      v4bf p;
      if (AMODE == 2) {
        p = *(const v4bf*)(Ab + (long long)mc * K + (k0 + col));
      } else {
        long long idx;
        if (AMODE == 0) {
          int gr = rowmap[mc];
          idx = (long long)gr * K + (k0 + col);
        } else {
          int kk = k0 + col;                  // quad never straddles the H_ boundary
          int gr = (kk < H_) ? r0t[Moff + mc] : r1t[Moff + mc];
          idx = (long long)gr * H_ + (kk & (H_ - 1));
        }
        v4f v = *(const v4f*)(Af + idx);
        p[0] = (bf16_t)v[0]; p[1] = (bf16_t)v[1];
        p[2] = (bf16_t)v[2]; p[3] = (bf16_t)v[3];
      }
      *(v4bf*)&As[row][col] = p;
    }
    // stage B tile (128 n-rows x 32 k)
    #pragma unroll
    for (int i = 0; i < 4; ++i) {
      int g = i * 256 + t;                    // 1024 quad groups
      int row = g >> 3;
      int col = (g & 7) * 4;
      int n = bn + row;
      v4bf p;
      if (n < N) {
        v4f v = *(const v4f*)(Bf + (long long)n * K + (k0 + col));
        p[0] = (bf16_t)v[0]; p[1] = (bf16_t)v[1];
        p[2] = (bf16_t)v[2]; p[3] = (bf16_t)v[3];
      } else {
        p[0] = (bf16_t)0.f; p[1] = (bf16_t)0.f; p[2] = (bf16_t)0.f; p[3] = (bf16_t)0.f;
      }
      *(v4bf*)&Bs[row][col] = p;
    }
    __syncthreads();

    // A fragments: 16-bit A 16x32 layout (ISA 7.12.2): elem i -> k=(i<8?i:i+8)+hf*8
    union { unsigned u[8]; v16bf v; } ua0, ua1, ub0, ub1;
    int mm0 = wrow * 32 + (lane & 15);
    const unsigned* arow0 = (const unsigned*)&As[mm0][0];
    const unsigned* arow1 = (const unsigned*)&As[mm0 + 16][0];
    #pragma unroll
    for (int j = 0; j < 4; ++j) {
      ua0.u[j]     = arow0[hf * 4 + j];
      ua0.u[4 + j] = arow0[8 + hf * 4 + j];
      ua1.u[j]     = arow1[hf * 4 + j];
      ua1.u[4 + j] = arow1[8 + hf * 4 + j];
    }
    // B fragments: 32x16 layout: elem i -> k=(i<8?i:i+8)+hf*8 (same as A)
    int nn0 = wcol * 32 + (lane & 15);
    const unsigned* brow0 = (const unsigned*)&Bs[nn0][0];
    const unsigned* brow1 = (const unsigned*)&Bs[nn0 + 16][0];
    #pragma unroll
    for (int j = 0; j < 4; ++j) {
      ub0.u[j]     = brow0[hf * 4 + j];
      ub0.u[4 + j] = brow0[8 + hf * 4 + j];
      ub1.u[j]     = brow1[hf * 4 + j];
      ub1.u[4 + j] = brow1[8 + hf * 4 + j];
    }
    acc[0][0] = __builtin_amdgcn_wmma_f32_16x16x32_bf16(false, ua0.v, false, ub0.v, (short)0, acc[0][0], false, false);
    acc[0][1] = __builtin_amdgcn_wmma_f32_16x16x32_bf16(false, ua0.v, false, ub1.v, (short)0, acc[0][1], false, false);
    acc[1][0] = __builtin_amdgcn_wmma_f32_16x16x32_bf16(false, ua1.v, false, ub0.v, (short)0, acc[1][0], false, false);
    acc[1][1] = __builtin_amdgcn_wmma_f32_16x16x32_bf16(false, ua1.v, false, ub1.v, (short)0, acc[1][1], false, false);
    __syncthreads();
  }

  // store: C/D 16x16 layout -> VGPR r: m = r + hf*8, n = lane&15
  size_t gmbase = 0;
  if (GMOUT) {
    size_t nm = (size_t)counts[0];
    gmbase = nm * (size_t)VOCAB_ + 2 * nm;    // after lm_logits, lm_labels, lm_loss
  }
  #pragma unroll
  for (int s = 0; s < 2; ++s) {
    #pragma unroll
    for (int u = 0; u < 2; ++u) {
      int n = bn + wcol * 32 + u * 16 + (lane & 15);
      if (n >= N) continue;
      float bv = bias ? bias[n] : 0.0f;
      #pragma unroll
      for (int r = 0; r < 8; ++r) {
        int m = bm + wrow * 32 + s * 16 + r + hf * 8;
        if (m < M) {
          float val = acc[s][u][r] + bv;
          if (GMOUT) Cp[gmbase + (size_t)(Moff + m) * EVOCAB_ + n] = val;
          else       Cp[(long long)m * ldc + n] = val;
        }
      }
    }
  }
}

// ---------------- gelu(exact) + LayerNorm, writes bf16 for the next GEMM ----------------
__global__ __launch_bounds__(256)
void k_head(const float* __restrict__ h, bf16_t* __restrict__ lnout,
            const float* __restrict__ g, const float* __restrict__ b,
            const int* __restrict__ counts, int mIdx, int Moff) {
  int M = counts[mIdx] - Moff;
  if (M > CAPR_) M = CAPR_;
  int row = blockIdx.x;
  if (row >= M) return;
  const float* hr = h + (long long)row * H_;
  int t = threadIdx.x;
  float loc[4]; float s = 0.f, s2 = 0.f;
  #pragma unroll
  for (int i = 0; i < 4; ++i) {
    float x = hr[t + i * 256];
    float ge = 0.5f * x * (1.0f + erff(x * 0.70710678118654752f));
    loc[i] = ge; s += ge; s2 += ge * ge;
  }
  __shared__ float rs[256], rq[256];
  rs[t] = s; rq[t] = s2; __syncthreads();
  for (int off = 128; off > 0; off >>= 1) {
    if (t < off) { rs[t] += rs[t + off]; rq[t] += rq[t + off]; }
    __syncthreads();
  }
  float mu  = rs[0] * (1.0f / H_);
  float var = rq[0] * (1.0f / H_) - mu * mu;
  float inv = rsqrtf(var + LN_EPS_);
  bf16_t* lo = lnout + (long long)row * H_;
  #pragma unroll
  for (int i = 0; i < 4; ++i) {
    int j = t + i * 256;
    lo[j] = (bf16_t)((loc[i] - mu) * inv * g[j] + b[j]);
  }
}

// ---------------- LM cross-entropy (+ labels as float) ----------------
__global__ __launch_bounds__(256)
void k_lm_ce(float* __restrict__ out, const int* __restrict__ counts, const int* __restrict__ lab) {
  int Nm = counts[0];
  int row = blockIdx.x;
  if (row >= Nm) return;
  const float* lg = out + (size_t)row * VOCAB_;
  int t = threadIdx.x;
  __shared__ float red[256];
  float mx = -3.4e38f;
  for (int j = t; j < VOCAB_; j += 256) mx = fmaxf(mx, lg[j]);
  red[t] = mx; __syncthreads();
  for (int off = 128; off > 0; off >>= 1) { if (t < off) red[t] = fmaxf(red[t], red[t + off]); __syncthreads(); }
  mx = red[0]; __syncthreads();
  float sm = 0.f;
  for (int j = t; j < VOCAB_; j += 256) sm += expf(lg[j] - mx);
  red[t] = sm; __syncthreads();
  for (int off = 128; off > 0; off >>= 1) { if (t < off) red[t] += red[t + off]; __syncthreads(); }
  if (t == 0) {
    int lb = lab[row];
    float loss = mx + logf(red[0]) - lg[lb];
    size_t o1 = (size_t)Nm * VOCAB_;
    out[o1 + row] = (float)lb;        // lm_labels
    out[o1 + Nm + row] = loss;        // lm_loss
  }
}

// ---------------- lasso |W| sum (deterministic 2-stage) ----------------
__global__ __launch_bounds__(256)
void k_abs_partial(const float* __restrict__ w, float* __restrict__ part, int n) {
  int t = blockIdx.x * 256 + threadIdx.x;
  float s = 0.f;
  for (int i = t; i < n; i += 256 * 256) s += fabsf(w[i]);
  __shared__ float red[256];
  red[threadIdx.x] = s; __syncthreads();
  for (int off = 128; off > 0; off >>= 1) { if (threadIdx.x < off) red[threadIdx.x] += red[threadIdx.x + off]; __syncthreads(); }
  if (threadIdx.x == 0) part[blockIdx.x] = red[0];
}
__global__ __launch_bounds__(256)
void k_abs_final(const float* __restrict__ part, float* __restrict__ osum) {
  __shared__ float red[256];
  red[threadIdx.x] = part[threadIdx.x]; __syncthreads();
  for (int off = 128; off > 0; off >>= 1) { if (threadIdx.x < off) red[threadIdx.x] += red[threadIdx.x + off]; __syncthreads(); }
  if (threadIdx.x == 0) osum[0] = red[0];
}

// ---------------- GM cross-entropy over 64 logits (+ labels, + lasso) ----------------
__global__ __launch_bounds__(256)
void k_gm_ce(float* __restrict__ out, const int* __restrict__ counts,
             const int* __restrict__ lab, const float* __restrict__ lasso) {
  int Nm = counts[0], Ne = counts[3];
  int row = blockIdx.x * 4 + (threadIdx.x >> 6);
  int l = threadIdx.x & 63;
  int base = (threadIdx.x >> 6) << 6;
  size_t o3 = (size_t)Nm * VOCAB_ + 2 * (size_t)Nm;
  bool active = (row < Ne);
  const float* lg = out + o3 + (size_t)(active ? row : 0) * EVOCAB_;
  __shared__ float red[256];
  float x = active ? lg[l] : -3.4e38f;
  red[threadIdx.x] = x; __syncthreads();
  for (int off = 32; off > 0; off >>= 1) {
    if (l < off) red[threadIdx.x] = fmaxf(red[threadIdx.x], red[threadIdx.x + off]);
    __syncthreads();
  }
  float mx = red[base]; __syncthreads();
  float e = active ? expf(x - mx) : 0.f;
  red[threadIdx.x] = e; __syncthreads();
  for (int off = 32; off > 0; off >>= 1) {
    if (l < off) red[threadIdx.x] += red[threadIdx.x + off];
    __syncthreads();
  }
  float sm = red[base];
  if (active && l == 0) {
    int lb = lab[row];
    float loss = mx + logf(sm) - lg[lb] + LASSO_ * lasso[0];
    out[o3 + (size_t)Ne * EVOCAB_ + row] = loss;            // gm_loss
    out[o3 + (size_t)Ne * EVOCAB_ + Ne + row] = (float)lb;  // edge_labels
  }
}

// ---------------- host ----------------
extern "C" void kernel_launch(void* const* d_in, const int* in_sizes, int n_in,
                              void* d_out, int out_size, void* d_ws, size_t ws_size,
                              hipStream_t stream) {
  (void)in_sizes; (void)n_in; (void)out_size; (void)ws_size;
  const float* hidden  = (const float*)d_in[0];
  const float* ebd     = (const float*)d_in[1];
  const float* eebd    = (const float*)d_in[2];
  const float* lm_w    = (const float*)d_in[3];
  const float* lm_b    = (const float*)d_in[4];
  const float* lm_g    = (const float*)d_in[5];
  const float* lm_bb   = (const float*)d_in[6];
  const float* lm_bias = (const float*)d_in[7];
  const float* gm_w    = (const float*)d_in[8];
  const float* gm_b    = (const float*)d_in[9];
  const float* gm_g    = (const float*)d_in[10];
  const float* gm_bb   = (const float*)d_in[11];
  const float* gm_bias = (const float*)d_in[12];
  const int* tflat     = (const int*)d_in[13];
  const int* adj       = (const int*)d_in[14];
  float* out = (float*)d_out;

  char* wsb = (char*)d_ws;
  size_t off = 0;
  auto alloc = [&](size_t bytes) { size_t o = off; off = (off + bytes + 255) & ~(size_t)255; return o; };
  int*    counts   = (int*)(wsb + alloc(64 * 4));
  float*  lassoSum = (float*)(wsb + alloc(4));
  float*  part     = (float*)(wsb + alloc(256 * 4));
  int*    mask_idx = (int*)(wsb + alloc((size_t)CAPR_ * 4));
  int*    lm_lab   = (int*)(wsb + alloc((size_t)CAPR_ * 4));
  int*    blkCnt   = (int*)(wsb + alloc(1024 * 4));
  int*    blkOff   = (int*)(wsb + alloc(1024 * 4));
  int*    e_b      = (int*)(wsb + alloc((size_t)ECAP_ * 4));
  int*    e_i      = (int*)(wsb + alloc((size_t)ECAP_ * 4));
  int*    e_j      = (int*)(wsb + alloc((size_t)ECAP_ * 4));
  int*    r0       = (int*)(wsb + alloc((size_t)E2CAP_ * 4));
  int*    r1       = (int*)(wsb + alloc((size_t)E2CAP_ * 4));
  int*    elab     = (int*)(wsb + alloc((size_t)E2CAP_ * 4));
  float*  buf_h    = (float*)(wsb + alloc((size_t)CAPR_ * H_ * 4));
  bf16_t* buf_ln   = (bf16_t*)(wsb + alloc((size_t)CAPR_ * H_ * 2));

  // index building (deterministic, ordered)
  k_compact_mask<<<1, 1024, 0, stream>>>(tflat, mask_idx, lm_lab, counts);
  k_edge_count<<<1024, 256, 0, stream>>>(adj, blkCnt);
  k_edge_scan<<<1, 1024, 0, stream>>>(blkCnt, blkOff, counts);
  k_edge_emit<<<1024, 256, 0, stream>>>(adj, blkOff, e_b, e_i, e_j, r0, r1, elab);
  k_trans_emit<<<1, 1024, 0, stream>>>(adj, e_b, e_i, e_j, r0, r1, elab, counts);

  // lasso scalar
  k_abs_partial<<<256, 256, 0, stream>>>(gm_w, part, H_ * 2 * H_);
  k_abs_final<<<1, 256, 0, stream>>>(part, lassoSum);

  // ---- LM head ----
  dim3 gdense(H_ / 128, CAPR_ / 64);                // 8 x 128
  k_gemm<0, false><<<gdense, 256, 0, stream>>>(hidden, nullptr, lm_w, lm_b,
      buf_h, H_, mask_idx, nullptr, nullptr, counts, 0, 0, H_, H_);
  k_head<<<CAPR_, 256, 0, stream>>>(buf_h, buf_ln, lm_g, lm_bb, counts, 0, 0);
  dim3 glog((VOCAB_ + 127) / 128, CAPR_ / 64);      // 393 x 128
  k_gemm<2, false><<<glog, 256, 0, stream>>>(nullptr, buf_ln, ebd, lm_bias,
      out, VOCAB_, nullptr, nullptr, nullptr, counts, 0, 0, VOCAB_, H_);
  k_lm_ce<<<CAPR_, 256, 0, stream>>>(out, counts, lm_lab);

  // ---- GM head, chunked over edge rows ----
  for (int c = 0; c < 5; ++c) {
    int Moff = c * CAPR_;
    k_gemm<1, false><<<gdense, 256, 0, stream>>>(hidden, nullptr, gm_w, gm_b,
        buf_h, H_, nullptr, r0, r1, counts, 3, Moff, H_, 2 * H_);
    k_head<<<CAPR_, 256, 0, stream>>>(buf_h, buf_ln, gm_g, gm_bb, counts, 3, Moff);
    dim3 gl2(1, CAPR_ / 64);
    k_gemm<2, true><<<gl2, 256, 0, stream>>>(nullptr, buf_ln, eebd, gm_bias,
        out, EVOCAB_, nullptr, nullptr, nullptr, counts, 3, Moff, EVOCAB_, H_);
  }
  k_gm_ce<<<E2CAP_ / 4, 256, 0, stream>>>(out, counts, elab, lassoSum);
}
